// MSARowAttentionWithPairBias_40501541601319
// MI455X (gfx1250) — matrix operand order repaired and checked by reading, not compiled, not run on hardware
//
#include <hip/hip_runtime.h>
#include <hip/hip_bf16.h>

typedef __attribute__((ext_vector_type(16))) __bf16 v16bf;
typedef __attribute__((ext_vector_type(8)))  float  v8f;

#define EPS 1e-5f

// ---------- helpers ----------
__device__ __forceinline__ unsigned short f2bf(float f) {
  union { float f; unsigned u; } x; x.f = f;
  unsigned u = x.u;
  u += 0x7fffu + ((u >> 16) & 1u);          // round-to-nearest-even
  return (unsigned short)(u >> 16);
}
__device__ __forceinline__ unsigned pack2(float a, float b) {
  return (unsigned)f2bf(a) | ((unsigned)f2bf(b) << 16);
}
__device__ __forceinline__ float bf2f(unsigned short u) {
  union { unsigned u; float f; } x; x.u = ((unsigned)u) << 16;
  return x.f;
}
// A-fragment (16x32 bf16): K index held in dword v, lane-half `half`
__device__ __forceinline__ int kA(int v, int half) {
  return (v < 4 ? 2 * v : 16 + 2 * (v - 4)) + half * 8;
}

// ---------- kernel 1: swizzle 5 weight matrices [256x256] f32 -> bf16 B-frag layout ----------
__global__ __launch_bounds__(256) void wswz(const float* Wq, const float* Wk,
                                            const float* Wv, const float* Wg,
                                            const float* Wo, unsigned* Wsw) {
  int gid = blockIdx.x * 256 + threadIdx.x;          // 0..163839
  int mat = gid >> 15;
  int idx = gid & 32767;
  const float* W = (mat == 0) ? Wq : (mat == 1) ? Wk : (mat == 2) ? Wv
                 : (mat == 3) ? Wg : Wo;
  int kcnt = idx >> 8;                               // kc*16 + nt
  int kc = kcnt >> 4, nt = kcnt & 15;
  int lv = idx & 255;
  int tl = lv >> 3, v = lv & 7;                      // target lane, dword
  int half = tl >> 4, nn = tl & 15;
  int K = kc * 32 + kA(v, half);                     // B-frag: same K order as A-frag
  int N = nt * 16 + nn;
  Wsw[gid] = pack2(W[K * 256 + N], W[(K + 1) * 256 + N]);
}

// ---------- kernel 2: LayerNorm(m) -> bf16 A-fragment layout ----------
__global__ __launch_bounds__(256) void ln_m_swz(const float* m, const float* g,
                                                const float* b, unsigned* mnsw) {
  __shared__ float X[16][257];
  __shared__ float MU[16], RS[16];
  int t = threadIdx.x;
  int rt = blockIdx.x;                               // row tile (16 rows of 65536)
  const float* src = m + (size_t)rt * 16 * 256;
  for (int s = 0; s < 16; ++s) X[s][t] = src[s * 256 + t];
  __syncthreads();
  int row = t >> 4, l = t & 15;                      // 16 threads per row
  float p = 0.f;
  for (int k = 0; k < 16; ++k) p += X[row][l + 16 * k];
  for (int mm = 8; mm >= 1; mm >>= 1) p += __shfl_xor(p, mm, 32);
  float mu = p * (1.f / 256.f);
  float q = 0.f;
  for (int k = 0; k < 16; ++k) { float d = X[row][l + 16 * k] - mu; q += d * d; }
  for (int mm = 8; mm >= 1; mm >>= 1) q += __shfl_xor(q, mm, 32);
  float rs = rsqrtf(q * (1.f / 256.f) + EPS);
  if (l == 0) { MU[row] = mu; RS[row] = rs; }
  __syncthreads();
  int tl = t >> 3, v = t & 7;                        // target lane, dword
  int mrow = tl & 15, half = tl >> 4;
  unsigned* dst = mnsw + (size_t)rt * 2048;
  for (int kc = 0; kc < 8; ++kc) {
    int K = kc * 32 + kA(v, half);
    float a = (X[mrow][K]     - MU[mrow]) * RS[mrow] * g[K]     + b[K];
    float c = (X[mrow][K + 1] - MU[mrow]) * RS[mrow] * g[K + 1] + b[K + 1];
    dst[kc * 256 + t] = pack2(a, c);
  }
}

// ---------- kernel 3: pair bias = LN(z) @ Wb -> bias[h][i][j] f32 ----------
__global__ __launch_bounds__(256) void pbias(const float* z, const float* g,
                                             const float* b, const float* Wb,
                                             float* bias) {
  int lane = threadIdx.x & 31;                       // channel (C_Z=32)
  int wid = (blockIdx.x * 256 + threadIdx.x) >> 5;   // i*256+j, 0..65535
  float x = z[(size_t)wid * 32 + lane];
  float p = x;
  for (int mm = 16; mm >= 1; mm >>= 1) p += __shfl_xor(p, mm, 32);
  float mu = p * (1.f / 32.f);
  float d = x - mu;
  float q = d * d;
  for (int mm = 16; mm >= 1; mm >>= 1) q += __shfl_xor(q, mm, 32);
  float zn = d * rsqrtf(q * (1.f / 32.f) + EPS) * g[lane] + b[lane];
  for (int h = 0; h < 8; ++h) {
    float s = zn * Wb[lane * 8 + h];
    for (int mm = 16; mm >= 1; mm >>= 1) s += __shfl_xor(s, mm, 32);
    if (lane == 0) bias[(size_t)h * 65536 + wid] = s;
  }
}

// ---------- kernel 4: projections q,k,v,gate = mn @ W (WMMA bf16) ----------
__global__ __launch_bounds__(256) void proj(const unsigned* mnsw, const unsigned* Wsw,
                                            const float* bg, unsigned* qsw,
                                            unsigned* ksw, unsigned* vsw,
                                            unsigned* gate) {
  __shared__ float T[8][16][17];
  int lane = threadIdx.x & 31;
  int w = threadIdx.x >> 5;
  int rt = blockIdx.x;                               // row tile
  int n = rt >> 4, it = rt & 15;
  int half = lane >> 4, nn = lane & 15;
  const unsigned* Abase = mnsw + (size_t)rt * 2048 + lane * 8;
  for (int tt = 0; tt < 8; ++tt) {
    int task = w * 8 + tt;                           // 64 tasks: mat(4) x ct(16)
    int mat = task >> 4, ct = task & 15;
    union { v16bf v; unsigned u[8]; } A, B;
    v8f acc = {0.f, 0.f, 0.f, 0.f, 0.f, 0.f, 0.f, 0.f};
    const unsigned* Bbase = Wsw + (size_t)mat * 32768 + ct * 256 + lane * 8;
    for (int kc = 0; kc < 8; ++kc) {
#pragma unroll
      for (int v = 0; v < 8; ++v) {
        A.u[v] = Abase[kc * 256 + v];
        B.u[v] = Bbase[kc * 4096 + v];
      }
      acc = __builtin_amdgcn_wmma_f32_16x16x32_bf16(false, A.v, false, B.v,
                                                    (short)0, acc, false, false);
    }
    // stage D-fragment to LDS with per-matrix transform
    float bgv = (mat == 3) ? bg[ct * 16 + nn] : 0.f;
#pragma unroll
    for (int r = 0; r < 8; ++r) {
      float d = acc[r];
      if (mat == 0) d *= 0.17677669529663687f;       // 1/sqrt(CH)
      if (mat == 3) d = 1.f / (1.f + __expf(-(d + bgv)));
      T[w][r + 8 * half][nn] = d;
    }
    __syncthreads();
    if (mat == 0) {                                  // q -> A-frag per (n,h,it)
      int h = ct >> 1, ctp = ct & 1;
      unsigned* dst = qsw + ((size_t)(n * 8 + h) * 16 + it) * 256 + lane * 8;
#pragma unroll
      for (int vv = 0; vv < 4; ++vv) {
        int v = ctp * 4 + vv;
        int cl = kA(v, half) - ctp * 16;             // channel-local 0..15
        dst[v] = pack2(T[w][nn][cl], T[w][nn][cl + 1]);
      }
    } else if (mat == 1) {                           // k -> B-frag (K=ch, N=j)
      int h = ct >> 1, hb = ct & 1;
      int t2 = nn + 16 * half;
      unsigned* dst = ksw + ((size_t)(n * 8 + h) * 16 + it) * 256 + t2 * 8;
#pragma unroll
      for (int vv = 0; vv < 4; ++vv) {
        int v = hb * 4 + vv;
        dst[v] = pack2(T[w][nn][half * 8 + 2 * vv], T[w][nn][half * 8 + 2 * vv + 1]);
      }
    } else if (mat == 2) {                           // v -> B-frag (K=j, N=ch)
      int h = ct >> 1, nt = ct & 1;
      int jc = it >> 1, hb = it & 1;
      int t2 = nn + 16 * hb;
      unsigned* dst = vsw + (((size_t)(n * 8 + h) * 8 + jc) * 2 + nt) * 256 + t2 * 8;
#pragma unroll
      for (int vv = 0; vv < 4; ++vv) {
        int v = half * 4 + vv;
        dst[v] = pack2(T[w][2 * v][nn], T[w][2 * v + 1][nn]);
      }
    } else {                                         // gate -> plain bf16 [NL][256]
      int rr = lane >> 1;
      size_t row = (size_t)rt * 16 + rr;
#pragma unroll
      for (int i2 = 0; i2 < 4; ++i2) {
        int cc2 = (lane & 1) * 4 + i2;
        gate[row * 128 + ct * 8 + cc2] =
            pack2(T[w][rr][cc2 * 2], T[w][rr][cc2 * 2 + 1]);
      }
    }
    __syncthreads();
  }
}

// ---------- kernel 5: attention per (n, h, i-tile), flash-style in LDS ----------
__global__ __launch_bounds__(32) void attn(const unsigned* qsw, const unsigned* ksw,
                                           const unsigned* vsw,
                                           const unsigned short* gate,
                                           const float* bias, const int* mask,
                                           unsigned* osw) {
  __shared__ float Lg[16][258];
  __shared__ float O[16][33];
  int lane = threadIdx.x;
  int id = blockIdx.x;
  int n = id >> 7, h = (id >> 4) & 7, it = id & 15;
  int i0 = it * 16;
  int half = lane >> 4, nn = lane & 15;
  union { v16bf v; unsigned u[8]; } A, B;
  const unsigned* qb = qsw + ((size_t)(n * 8 + h) * 16 + it) * 256 + lane * 8;
#pragma unroll
  for (int v = 0; v < 8; ++v) A.u[v] = qb[v];
  // logits = (q/sqrt(CH)) k^T + bias, masked over j
  for (int jt = 0; jt < 16; ++jt) {
    int j0 = jt * 16;
    v8f c;
#pragma unroll
    for (int r = 0; r < 8; ++r)
      c[r] = bias[(size_t)h * 65536 + (i0 + r + 8 * half) * 256 + j0 + nn];
    const unsigned* kb = ksw + ((size_t)(n * 8 + h) * 16 + jt) * 256 + lane * 8;
#pragma unroll
    for (int v = 0; v < 8; ++v) B.u[v] = kb[v];
    c = __builtin_amdgcn_wmma_f32_16x16x32_bf16(false, A.v, false, B.v,
                                                (short)0, c, false, false);
    int mk = mask[n * 256 + j0 + nn];
#pragma unroll
    for (int r = 0; r < 8; ++r)
      Lg[r + 8 * half][j0 + nn] = (mk == 0) ? -1e9f : c[r];
  }
  __syncthreads();
  // softmax over j=256: 2 lanes per row
  float mx = -3.4e38f;
  for (int t = 0; t < 128; ++t) mx = fmaxf(mx, Lg[nn][half * 128 + t]);
  mx = fmaxf(mx, __shfl_xor(mx, 16, 32));
  float s = 0.f;
  for (int t = 0; t < 128; ++t) {
    float e = __expf(Lg[nn][half * 128 + t] - mx);
    Lg[nn][half * 128 + t] = e;
    s += e;
  }
  s += __shfl_xor(s, 16, 32);
  float inv = 1.f / s;
  __syncthreads();
  // out = attn @ v  (K = j, 8 chunks of 32; N = ch, 2 tiles of 16)
  v8f acc0 = {0.f, 0.f, 0.f, 0.f, 0.f, 0.f, 0.f, 0.f};
  v8f acc1 = {0.f, 0.f, 0.f, 0.f, 0.f, 0.f, 0.f, 0.f};
  for (int jc = 0; jc < 8; ++jc) {
#pragma unroll
    for (int v = 0; v < 8; ++v) {
      int K = jc * 32 + 16 * half + 2 * v;
      A.u[v] = pack2(Lg[nn][K] * inv, Lg[nn][K + 1] * inv);
    }
    const unsigned* vb = vsw + (((size_t)(n * 8 + h) * 8 + jc) * 2) * 256 + lane * 8;
#pragma unroll
    for (int v = 0; v < 8; ++v) B.u[v] = vb[v];
    acc0 = __builtin_amdgcn_wmma_f32_16x16x32_bf16(false, A.v, false, B.v,
                                                   (short)0, acc0, false, false);
#pragma unroll
    for (int v = 0; v < 8; ++v) B.u[v] = vb[256 + v];
    acc1 = __builtin_amdgcn_wmma_f32_16x16x32_bf16(false, A.v, false, B.v,
                                                   (short)0, acc1, false, false);
  }
  // gate and re-layout to A-fragment for the output GEMM
#pragma unroll
  for (int r = 0; r < 8; ++r) {
    size_t grow = (size_t)n * 256 + i0 + r + 8 * half;
    float g0 = bf2f(gate[grow * 256 + h * 32 + nn]);
    float g1 = bf2f(gate[grow * 256 + h * 32 + 16 + nn]);
    O[r + 8 * half][nn] = acc0[r] * g0;
    O[r + 8 * half][16 + nn] = acc1[r] * g1;
  }
  __syncthreads();
  unsigned* dst = osw + (size_t)(n * 16 + it) * 2048 + h * 256 + lane * 8;
#pragma unroll
  for (int v = 0; v < 8; ++v) {
    int K = kA(v, half);
    dst[v] = pack2(O[nn][K], O[nn][K + 1]);
  }
}

// ---------- kernel 6: out = o @ Wo + bo, masked by msa_mask over rows ----------
__global__ __launch_bounds__(256) void outproj(const unsigned* osw, const unsigned* Wsw,
                                               const float* bo, const int* mask,
                                               float* out) {
  __shared__ float T[8][16][17];
  int lane = threadIdx.x & 31;
  int w = threadIdx.x >> 5;
  int rt = blockIdx.x;
  int half = lane >> 4, nn = lane & 15;
  const unsigned* Abase = osw + (size_t)rt * 2048 + lane * 8;
  const unsigned* WoB = Wsw + (size_t)4 * 32768;
  for (int tt = 0; tt < 2; ++tt) {
    int ct = w * 2 + tt;
    union { v16bf v; unsigned u[8]; } A, B;
    v8f acc = {0.f, 0.f, 0.f, 0.f, 0.f, 0.f, 0.f, 0.f};
    const unsigned* Bbase = WoB + ct * 256 + lane * 8;
    for (int kc = 0; kc < 8; ++kc) {
#pragma unroll
      for (int v = 0; v < 8; ++v) {
        A.u[v] = Abase[kc * 256 + v];
        B.u[v] = Bbase[kc * 4096 + v];
      }
      acc = __builtin_amdgcn_wmma_f32_16x16x32_bf16(false, A.v, false, B.v,
                                                    (short)0, acc, false, false);
    }
#pragma unroll
    for (int r = 0; r < 8; ++r) T[w][r + 8 * half][nn] = acc[r];
    __syncthreads();
    int rr = lane >> 1;
    size_t row = (size_t)rt * 16 + rr;
    float mk = (mask[row] != 0) ? 1.f : 0.f;
#pragma unroll
    for (int i2 = 0; i2 < 8; ++i2) {
      int col = ct * 16 + (lane & 1) * 8 + i2;
      out[row * 256 + col] = (T[w][rr][(lane & 1) * 8 + i2] + bo[col]) * mk;
    }
    __syncthreads();
  }
}

// ---------- launch ----------
extern "C" void kernel_launch(void* const* d_in, const int* in_sizes, int n_in,
                              void* d_out, int out_size, void* d_ws, size_t ws_size,
                              hipStream_t stream) {
  const float* m      = (const float*)d_in[0];
  const float* z      = (const float*)d_in[1];
  const int*   msk    = (const int*)d_in[2];
  const float* ln_m_g = (const float*)d_in[3];
  const float* ln_m_b = (const float*)d_in[4];
  const float* ln_z_g = (const float*)d_in[5];
  const float* ln_z_b = (const float*)d_in[6];
  const float* Wq     = (const float*)d_in[7];
  const float* Wk     = (const float*)d_in[8];
  const float* Wv     = (const float*)d_in[9];
  const float* Wb     = (const float*)d_in[10];
  const float* Wg     = (const float*)d_in[11];
  const float* bg     = (const float*)d_in[12];
  const float* Wo     = (const float*)d_in[13];
  const float* bo     = (const float*)d_in[14];
  float* out = (float*)d_out;

  char* ws = (char*)d_ws;
  const size_t SZ_W    = 5u * 32768u * 4u;       //   655,360 B: swizzled weights
  const size_t SZ_BIG  = 33554432u;              // 33.55 MB each
  unsigned* Wsw  = (unsigned*)(ws);
  unsigned* mnsw = (unsigned*)(ws + SZ_W);                 // aliased by osw
  unsigned* qsw  = (unsigned*)(ws + SZ_W + 1 * SZ_BIG);
  unsigned* ksw  = (unsigned*)(ws + SZ_W + 2 * SZ_BIG);
  unsigned* vsw  = (unsigned*)(ws + SZ_W + 3 * SZ_BIG);
  unsigned* gate = (unsigned*)(ws + SZ_W + 4 * SZ_BIG);
  float*    bias = (float*)   (ws + SZ_W + 5 * SZ_BIG);
  unsigned* osw  = mnsw;  // mn consumed by proj before attn writes o

  wswz<<<640, 256, 0, stream>>>(Wq, Wk, Wv, Wg, Wo, Wsw);
  ln_m_swz<<<4096, 256, 0, stream>>>(m, ln_m_g, ln_m_b, mnsw);
  pbias<<<8192, 256, 0, stream>>>(z, ln_z_g, ln_z_b, Wb, bias);
  proj<<<4096, 256, 0, stream>>>(mnsw, Wsw, bg, qsw, ksw, vsw, gate);
  attn<<<32768, 32, 0, stream>>>(qsw, ksw, vsw, (const unsigned short*)gate,
                                 bias, msk, osw);
  outproj<<<4096, 256, 0, stream>>>(osw, Wsw, bo, msk, out);
}
